// VectorQuantizer_48198122996387
// MI455X (gfx1250) — matrix operand order repaired and checked by reading, not compiled, not run on hardware
//
#include <hip/hip_runtime.h>
#include <hip/hip_bf16.h>
#include <cmath>

typedef __attribute__((ext_vector_type(16))) __bf16 v16bf;
typedef __attribute__((ext_vector_type(8)))  float  v8f;

#define K_CODES   1024
#define DIM       256
#define BROWS     65536
#define LDS_PITCH 264   // 256 bf16 + 8 pad -> 528B row pitch; b128 frag loads hit disjoint bank quads

__device__ __forceinline__ unsigned short f32_to_bf16(float f) {
    unsigned int u = __float_as_uint(f);
    u += 0x7FFFu + ((u >> 16) & 1u);   // round-to-nearest-even
    return (unsigned short)(u >> 16);
}

// ---------------------------------------------------------------------------
// Kernel 0: W f32 -> bf16, wnorm[k] = sum W[k]^2 (exact f32), zero accumulators
// ---------------------------------------------------------------------------
__global__ __launch_bounds__(256) void vq_prep(const float* __restrict__ W,
                                               unsigned short* __restrict__ Wb,
                                               float* __restrict__ wnorm,
                                               int* __restrict__ counts,
                                               float* __restrict__ sumsq) {
    const int k = blockIdx.x;
    const int t = threadIdx.x;
    float w = W[k * DIM + t];
    Wb[k * DIM + t] = f32_to_bf16(w);
    float s = w * w;
    #pragma unroll
    for (int m = 16; m >= 1; m >>= 1) s += __shfl_xor(s, m, 32);
    __shared__ float red[8];
    if ((t & 31) == 0) red[t >> 5] = s;
    __syncthreads();
    if (t == 0) {
        float tot = 0.f;
        #pragma unroll
        for (int i = 0; i < 8; ++i) tot += red[i];
        wnorm[k] = tot;
        counts[k] = 0;
        if (k == 0) *sumsq = 0.f;
    }
}

// ---------------------------------------------------------------------------
// Kernel 1: distance GEMM via v_wmma_f32_16x16x32_bf16 + per-row argmin.
//   block = 256 threads (8 waves), 128 rows of x, all 1024 codes.
//   W tiles (16 codes x 256 D bf16 = 8KB) double-buffered in LDS via
//   GLOBAL_LOAD_ASYNC_TO_LDS_B128 (ASYNCcnt), shared by all 8 waves.
// ---------------------------------------------------------------------------
__device__ __forceinline__ void stage_wtile_async(const unsigned short* __restrict__ Wb,
                                                  unsigned short* wbuf, int tile, int tid) {
    // 8KB tile = 512 x 16B chunks; 256 threads issue 2 async b128 copies each.
    #pragma unroll
    for (int p = 0; p < 2; ++p) {
        const int j   = tid + p * 256;   // 0..511
        const int row = j >> 5;          // 16 rows
        const int c16 = j & 31;          // 32 16B chunks per row
        const unsigned short* g = Wb + (size_t)(tile * 16 + row) * DIM + c16 * 8;
        const unsigned lds_off = (unsigned)(uintptr_t)(wbuf + row * LDS_PITCH + c16 * 8);
        asm volatile("global_load_async_to_lds_b128 %0, %1, off"
                     :: "v"(lds_off), "v"(g) : "memory");
    }
}

__device__ __forceinline__ void wait_async0() {
    asm volatile("s_wait_asynccnt 0x0" ::: "memory");
}

__global__ __launch_bounds__(256) void vq_argmin(const float* __restrict__ X,
                                                 const unsigned short* __restrict__ Wb,
                                                 const float* __restrict__ wnorm,
                                                 int* __restrict__ idx) {
    __shared__ unsigned short xs[128 * LDS_PITCH];        // 66   KB bf16 x-tile
    __shared__ unsigned short wtile[2][16 * LDS_PITCH];   // 16.5 KB double-buffered W tile

    const int tid     = threadIdx.x;
    const int rowBase = blockIdx.x * 128;

    // Stage 128 rows of x into LDS as bf16 (coalesced float4 reads).
    for (int i = tid; i < 128 * 64; i += 256) {
        const int r  = i >> 6;
        const int c4 = i & 63;
        const float4 v = ((const float4*)X)[(size_t)(rowBase + r) * 64 + c4];
        ushort4 h;
        h.x = f32_to_bf16(v.x); h.y = f32_to_bf16(v.y);
        h.z = f32_to_bf16(v.z); h.w = f32_to_bf16(v.w);
        *(ushort4*)&xs[r * LDS_PITCH + c4 * 4] = h;
    }

    // Kick off async copy of the first W tile while x staging completes.
    stage_wtile_async(Wb, &wtile[0][0], 0, tid);
    wait_async0();
    __syncthreads();

    const int lane = tid & 31;
    const int wave = tid >> 5;
    const int half = lane >> 4;   // 0: lanes 0-15, 1: lanes 16-31
    const int nlan = lane & 15;

    union Frag { v16bf v; uint4 q[2]; };

    // Preload all 8 A-fragments (row m = wave*16 + nlan, full D=256).
    // 16-bit A layout: lanes 0-15 hold K {0..7, 16..23}; lanes 16-31 hold K {8..15, 24..31}.
    v16bf afr[8];
    {
        const unsigned short* xr = &xs[(wave * 16 + nlan) * LDS_PITCH];
        #pragma unroll
        for (int kb = 0; kb < 8; ++kb) {
            Frag a;
            a.q[0] = *(const uint4*)(xr + kb * 32 + half * 8);        // K = kb*32 + half*8 ..+7
            a.q[1] = *(const uint4*)(xr + kb * 32 + 16 + half * 8);   // K = kb*32+16+half*8 ..+7
            afr[kb] = a.v;
        }
    }

    float bestv[8];
    int   besti[8];
    #pragma unroll
    for (int r = 0; r < 8; ++r) { bestv[r] = 3.4e38f; besti[r] = 0; }

    // Sweep 64 tiles of 16 codes each; double-buffered async W staging.
    for (int t = 0; t < 64; ++t) {
        const int buf = t & 1;
        if (t < 63)
            stage_wtile_async(Wb, &wtile[buf ^ 1][0], t + 1, tid);

        const float wn = wnorm[t * 16 + nlan];
        // B layout (32x16 bf16): lanes 0-15 -> N=lane, K {0..7, 16..23}; lanes 16-31 -> N=lane-16, K {8..15, 24..31}.
        const unsigned short* wrow = &wtile[buf][nlan * LDS_PITCH + half * 8];

        v8f acc = {0.f, 0.f, 0.f, 0.f, 0.f, 0.f, 0.f, 0.f};
        #pragma unroll
        for (int kb = 0; kb < 8; ++kb) {
            Frag b;
            b.q[0] = *(const uint4*)(wrow + kb * 32);       // K = kb*32 + half*8 ..+7
            b.q[1] = *(const uint4*)(wrow + kb * 32 + 16);  // K = kb*32+16+half*8 ..+7
            acc = __builtin_amdgcn_wmma_f32_16x16x32_bf16(
                      false, afr[kb], false, b.v, (short)0, acc, false, false);
        }

        // C layout: VGPR r, lane -> row (r + 8*half), col nlan. Candidate code id:
        const int cand = t * 16 + nlan;
        #pragma unroll
        for (int r = 0; r < 8; ++r) {
            float d = wn - 2.0f * acc[r];
            bool better = d < bestv[r];          // strict '<' keeps earliest (lowest) index
            bestv[r] = better ? d : bestv[r];
            besti[r] = better ? cand : besti[r];
        }

        // All waves done with wtile[buf]; their next-tile asyncs must land before reuse.
        wait_async0();
        __syncthreads();
    }

    // 16-lane min+index reduction per C register (xor masks 1..8 stay within halves).
    #pragma unroll
    for (int r = 0; r < 8; ++r) {
        float v  = bestv[r];
        int   bi = besti[r];
        #pragma unroll
        for (int m = 1; m < 16; m <<= 1) {
            float ov = __shfl_xor(v, m, 32);
            int   oi = __shfl_xor(bi, m, 32);
            if (ov < v || (ov == v && oi < bi)) { v = ov; bi = oi; }
        }
        if (nlan == 0)
            idx[rowBase + wave * 16 + half * 8 + r] = bi;
    }
}

// ---------------------------------------------------------------------------
// Kernel 2: gather W[idx] -> out (f32), sum (q-x)^2, histogram counts.
// ---------------------------------------------------------------------------
__global__ __launch_bounds__(256) void vq_gather(const float* __restrict__ X,
                                                 const float* __restrict__ W,
                                                 const int* __restrict__ idx,
                                                 float* __restrict__ out,
                                                 int* __restrict__ counts,
                                                 float* __restrict__ sumsq) {
    const int e4  = blockIdx.x * 256 + threadIdx.x;   // float4 index, total B*D/4
    const int row = e4 >> 6;
    const int c4  = e4 & 63;
    const int ci  = idx[row];

    const float4 q = ((const float4*)W)[(size_t)ci * 64 + c4];
    const float4 x = ((const float4*)X)[(size_t)e4];
    ((float4*)out)[(size_t)e4] = q;

    const float dx = q.x - x.x, dy = q.y - x.y, dz = q.z - x.z, dw = q.w - x.w;
    float s = dx * dx + dy * dy + dz * dz + dw * dw;

    if (c4 == 0) atomicAdd(&counts[ci], 1);

    #pragma unroll
    for (int m = 16; m >= 1; m >>= 1) s += __shfl_xor(s, m, 32);
    __shared__ float red[8];
    if ((threadIdx.x & 31) == 0) red[threadIdx.x >> 5] = s;
    __syncthreads();
    if (threadIdx.x == 0) {
        float tot = 0.f;
        #pragma unroll
        for (int i = 0; i < 8; ++i) tot += red[i];
        atomicAdd(sumsq, tot);
    }
}

// ---------------------------------------------------------------------------
// Kernel 3: loss + perplexity into the output tail.
// ---------------------------------------------------------------------------
__global__ __launch_bounds__(1024) void vq_final(const int* __restrict__ counts,
                                                 const float* __restrict__ sumsq,
                                                 float* __restrict__ out_tail) {
    const int t = threadIdx.x;
    float p = (float)counts[t] * (1.0f / 65536.0f);
    float term = -p * logf(p + 1e-10f);
    #pragma unroll
    for (int m = 16; m >= 1; m >>= 1) term += __shfl_xor(term, m, 32);
    __shared__ float red[32];
    if ((t & 31) == 0) red[t >> 5] = term;
    __syncthreads();
    if (t == 0) {
        float tot = 0.f;
        #pragma unroll
        for (int i = 0; i < 32; ++i) tot += red[i];
        out_tail[0] = (*sumsq) * 1.25f / (65536.0f * 256.0f);  // q_loss + 0.25*e_loss
        out_tail[1] = expf(tot);                               // perplexity
    }
}

// ---------------------------------------------------------------------------
extern "C" void kernel_launch(void* const* d_in, const int* in_sizes, int n_in,
                              void* d_out, int out_size, void* d_ws, size_t ws_size,
                              hipStream_t stream) {
    const float* X = (const float*)d_in[0];   // (65536, 256) f32
    const float* W = (const float*)d_in[1];   // (1024, 256)  f32
    float* out = (float*)d_out;               // 65536*256 quantized + loss + perplexity

    // Workspace carve-up (~776 KB total)
    unsigned short* Wb    = (unsigned short*)d_ws;            // 1024*256 bf16 = 512 KB
    float*          wnorm = (float*)(Wb + K_CODES * DIM);     // 4 KB
    int*            idx   = (int*)(wnorm + K_CODES);          // 256 KB
    int*            counts= idx + BROWS;                      // 4 KB
    float*          sumsq = (float*)(counts + K_CODES);       // 4 B

    vq_prep  <<<K_CODES,            256, 0, stream>>>(W, Wb, wnorm, counts, sumsq);
    vq_argmin<<<BROWS / 128,        256, 0, stream>>>(X, Wb, wnorm, idx);
    vq_gather<<<BROWS * DIM / 1024, 256, 0, stream>>>(X, W, idx, out, counts, sumsq);
    vq_final <<<1,                 1024, 0, stream>>>(counts, sumsq, out + (size_t)BROWS * DIM);
}
